// NodeEdgeBlock_78116865179717
// MI455X (gfx1250) — compile-verified, matrix operand order checked by reading, not run on hardware
//
#include <hip/hip_runtime.h>

typedef __attribute__((ext_vector_type(16))) __bf16 v16bf;
typedef __attribute__((ext_vector_type(8)))  float  v8f;

#define BSZ 16
#define NN  128
#define DXC 256
#define DEC 64

// ---------------------------------------------------------------------------
// WMMA fragment index helper (16x16x32 bf16 layout, wave32):
//  A (MxK tile 16x32): lane = m%16 + 16*hi(k'), elems: k'(0..7)->0..7, 8..15->0..7(hi),
//  16..23->8..15, 24..31->8..15(hi). Same mapping used for B with n as the 16-dim.
// ---------------------------------------------------------------------------
__device__ __forceinline__ int frag_off(int ktiles, int t16, int kt, int i16, int k32) {
    int hi   = (k32 >> 3) & 1;
    int elem = (k32 & 7) | ((k32 & 16) ? 8 : 0);
    int lane = i16 + (hi << 4);
    return ((t16 * ktiles + kt) * 32 + lane) * 16 + elem;
}

__device__ __forceinline__ unsigned fenc(float f) {
    unsigned u = __float_as_uint(f);
    return (u & 0x80000000u) ? ~u : (u | 0x80000000u);
}
__device__ __forceinline__ float fdec(unsigned u) {
    return __uint_as_float((u & 0x80000000u) ? (u ^ 0x80000000u) : ~u);
}

// ---------------------------------------------------------------------------
// Pack a row-major f32 weight (K x Ncols) into bf16 WMMA fragment order.
// ---------------------------------------------------------------------------
__global__ void pack_w(const float* __restrict__ W, __bf16* __restrict__ out, int K, int Ncols) {
    int idx = blockIdx.x * blockDim.x + threadIdx.x;
    if (idx >= K * Ncols) return;
    int k = idx / Ncols, n = idx % Ncols;
    out[frag_off(K >> 5, n >> 4, k >> 5, n & 15, k & 31)] = (__bf16)W[idx];
}

__global__ void init_stats(float* zsum, float* zsq, unsigned* zmin, unsigned* zmax) {
    int i = blockIdx.x * blockDim.x + threadIdx.x;
    if (i < BSZ * DEC) { zsum[i] = 0.f; zsq[i] = 0.f; zmin[i] = 0xFFFFFFFFu; zmax[i] = 0u; }
}

// ---------------------------------------------------------------------------
// Q/K/V projections: 64 rows per block, WMMA bf16, masked by node_mask.
// ---------------------------------------------------------------------------
__global__ __launch_bounds__(256) void node_proj(
    const float* __restrict__ X, const int* __restrict__ nm,
    const __bf16* __restrict__ pWq, const __bf16* __restrict__ pWk, const __bf16* __restrict__ pWv,
    const float* __restrict__ bq, const float* __restrict__ bk, const float* __restrict__ bv,
    float* __restrict__ Qo, float* __restrict__ Ko, float* __restrict__ Vo)
{
    __shared__ __align__(32) __bf16 sA[64 * 256];
    const int row0 = blockIdx.x * 64;
    const int tid = threadIdx.x;
    for (int e = tid; e < 64 * 256; e += 256) {
        int r = e >> 8, c = e & 255;
        sA[frag_off(8, r >> 4, c >> 5, r & 15, c & 31)] = (__bf16)X[(size_t)(row0 + r) * DXC + c];
    }
    __syncthreads();
    const int wv = tid >> 5, lane = tid & 31, lh = lane >> 4, l16 = lane & 15;
    for (int tI = wv; tI < 192; tI += 8) {
        int mat = tI >> 6, t = tI & 63, ms = t >> 4, nt = t & 15;
        const __bf16* Wp = (mat == 0) ? pWq : ((mat == 1) ? pWk : pWv);
        const float*  bp = (mat == 0) ? bq  : ((mat == 1) ? bk  : bv);
        float*        Op = (mat == 0) ? Qo  : ((mat == 1) ? Ko  : Vo);
        v8f acc = {};
#pragma unroll
        for (int kt = 0; kt < 8; ++kt) {
            v16bf af = *(const v16bf*)(sA + ((ms * 8 + kt) * 32 + lane) * 16);
            v16bf bf = *(const v16bf*)(Wp + ((nt * 8 + kt) * 32 + lane) * 16);
            acc = __builtin_amdgcn_wmma_f32_16x16x32_bf16(false, af, false, bf, (short)0, acc, false, false);
        }
        int c = nt * 16 + l16;
        float bb = bp[c];
#pragma unroll
        for (int r = 0; r < 8; ++r) {
            int grow = row0 + ms * 16 + lh * 8 + r;
            Op[(size_t)grow * DXC + c] = (acc[r] + bb) * (float)nm[grow];
        }
    }
}

// ---------------------------------------------------------------------------
// FiLM vectors from y: yE_mul, yE_add, yX_mul, yX_add  (16 x 256 each)
// ---------------------------------------------------------------------------
__global__ void ymix(const float* __restrict__ y,
                     const float* Wyem, const float* byem, const float* Wyea, const float* byea,
                     const float* Wyxm, const float* byxm, const float* Wyxa, const float* byxa,
                     float* yEm, float* yEa, float* yXm, float* yXa)
{
    __shared__ float sy[64];
    int b = blockIdx.x, t = threadIdx.x;
    if (t < 64) sy[t] = y[b * 64 + t];
    __syncthreads();
    float a0 = byem[t], a1 = byea[t], a2 = byxm[t], a3 = byxa[t];
    for (int k = 0; k < 64; ++k) {
        float yv = sy[k];
        a0 += yv * Wyem[k * 256 + t]; a1 += yv * Wyea[k * 256 + t];
        a2 += yv * Wyxm[k * 256 + t]; a3 += yv * Wyxa[k * 256 + t];
    }
    yEm[b * 256 + t] = a0; yEa[b * 256 + t] = a1;
    yXm[b * 256 + t] = a2; yXa[b * 256 + t] = a3;
}

// X pooling -> zx = [mean | min | max | std(ddof=1)]  (16 x 1024)
__global__ void x_pool(const float* __restrict__ X, float* __restrict__ zx) {
    int b = blockIdx.x, c = threadIdx.x;
    float s = 0.f, sq = 0.f, mn = 3.4e38f, mx = -3.4e38f;
    for (int j = 0; j < NN; ++j) {
        float v = X[((size_t)b * NN + j) * DXC + c];
        s += v; sq += v * v; mn = fminf(mn, v); mx = fmaxf(mx, v);
    }
    float mean = s / 128.f;
    float var = fmaxf((sq - 128.f * mean * mean) / 127.f, 0.f);
    zx[b * 1024 + c] = mean;
    zx[b * 1024 + 256 + c] = mn;
    zx[b * 1024 + 512 + c] = mx;
    zx[b * 1024 + 768 + c] = sqrtf(var);
}

// ---------------------------------------------------------------------------
// Fused edge kernel: one block per (b,i). Computes E1/E2 (WMMA), Y, online
// per-channel softmax over j with V accumulation, FiLM, e_out (WMMA), masked
// newE store, wV store, and fused E pooling atomics.
// ---------------------------------------------------------------------------
__global__ __launch_bounds__(256) void edge_kernel(
    const float* __restrict__ E, const int* __restrict__ nm,
    const float* __restrict__ Qb, const float* __restrict__ Kb, const float* __restrict__ Vb,
    const float* __restrict__ yEm, const float* __restrict__ yEa,
    const __bf16* __restrict__ pWem, const __bf16* __restrict__ pWea, const __bf16* __restrict__ pWeo,
    const float* __restrict__ bem, const float* __restrict__ bea, const float* __restrict__ beo,
    float* __restrict__ outE, float* __restrict__ wVo,
    float* __restrict__ zsum, float* __restrict__ zsq,
    unsigned* __restrict__ zminE, unsigned* __restrict__ zmaxE)
{
    __shared__ __align__(32) unsigned char smemRaw[64 * 64 * 2 + 64 * 256 * 2]; // 40 KB
    __shared__ float smask[NN];                       // node_mask row cache
    __bf16* sE = (__bf16*)smemRaw;                    // 64x64 E block, A-frag order
    __bf16* sP = (__bf16*)(smemRaw + 64 * 64 * 2);    // 64x256 FiLM(Y), A-frag order
    float*  sStat = (float*)smemRaw;                  // reused after GEMMs: m/l/acc partials

    const int tid = threadIdx.x;
    const int wv = tid >> 5, lane = tid & 31, lh = lane >> 4, l16 = lane & 15;
    const int stripe = wv >> 1, nhalf = wv & 1;
    const int bi = blockIdx.x, b = bi >> 7, i = bi & 127;
    const float xmi = (float)nm[b * NN + i];
    const float inv_s = 0.17677669529663687f; // 1/sqrt(32)
    const float NEGBIG = -3.4e38f;

    if (tid < NN) smask[tid] = (float)nm[b * NN + tid];

    float st_m[8], st_l[8], st_a[8];
#pragma unroll
    for (int t = 0; t < 8; ++t) { st_m[t] = NEGBIG; st_l[t] = 0.f; st_a[t] = 0.f; }

    const int pc = tid & 63, pg = tid >> 6;   // E pooling: channel pc, row group pg
    float psum = 0.f, psq = 0.f, pmn = 3.4e38f, pmx = -3.4e38f;

    const float* Erow = E + (((size_t)b * NN + i) * NN) * DEC;

    for (int jb = 0; jb < 2; ++jb) {
        // --- load E block (64 rows x 64 ch), fuse pooling, store bf16 A-frags ---
#pragma unroll
        for (int r = 0; r < 16; ++r) {
            int j = pg * 16 + r;
            float v = Erow[(size_t)(jb * 64 + j) * DEC + pc];
            psum += v; psq += v * v; pmn = fminf(pmn, v); pmx = fmaxf(pmx, v);
            sE[frag_off(2, j >> 4, pc >> 5, j & 15, pc & 31)] = (__bf16)v;
        }
        if (jb == 0) {
            // prefetch the second 64-row E block (16 KB) while computing this one
            __builtin_prefetch((const char*)(Erow + (size_t)64 * DEC) + tid * 64, 0, 3);
        }
        __syncthreads();

        // --- phase 1: E1/E2 GEMMs + Y + online softmax + FiLM -> sP ---
#pragma unroll
        for (int tt = 0; tt < 8; ++tt) {
            int nt = nhalf * 8 + tt;
            v8f a1 = {}, a2 = {};
#pragma unroll
            for (int kt = 0; kt < 2; ++kt) {
                v16bf af = *(const v16bf*)(sE + ((stripe * 2 + kt) * 32 + lane) * 16);
                v16bf b1 = *(const v16bf*)(pWem + ((nt * 2 + kt) * 32 + lane) * 16);
                v16bf b2 = *(const v16bf*)(pWea + ((nt * 2 + kt) * 32 + lane) * 16);
                a1 = __builtin_amdgcn_wmma_f32_16x16x32_bf16(false, af, false, b1, (short)0, a1, false, false);
                a2 = __builtin_amdgcn_wmma_f32_16x16x32_bf16(false, af, false, b2, (short)0, a2, false, false);
            }
            int ch = nt * 16 + l16;
            float be1 = bem[ch], be2 = bea[ch];
            float qv = Qb[((size_t)b * NN + i) * DXC + ch] * inv_s;
            float fm = yEm[b * DXC + ch] + 1.0f;
            float fa = yEa[b * DXC + ch];
#pragma unroll
            for (int r = 0; r < 8; ++r) {
                int jl = stripe * 16 + lh * 8 + r;
                int j = jb * 64 + jl;
                float mjf = smask[j];
                float em = xmi * mjf;
                float e1 = (a1[r] + be1) * em;
                float e2 = (a2[r] + be2) * em;
                float kv = Kb[((size_t)b * NN + j) * DXC + ch];
                float Yv = qv * kv * (e1 + 1.0f) + e2;
                // branchless masked online softmax over keys + attn*V accumulation
                float vv = Vb[((size_t)b * NN + j) * DXC + ch];
                float Yeff = (mjf > 0.f) ? Yv : NEGBIG;
                float mo = st_m[tt];
                float mnw = fmaxf(mo, Yeff);
                float sc = __expf(mo - mnw);
                float p = (mjf > 0.f) ? __expf(Yv - mnw) : 0.f;
                st_l[tt] = st_l[tt] * sc + p;
                st_a[tt] = st_a[tt] * sc + p * vv;
                st_m[tt] = mnw;
                sP[frag_off(8, stripe, ch >> 5, jl & 15, ch & 31)] = (__bf16)(fa + fm * Yv);
            }
        }
        __syncthreads();

        // --- phase 2: newE = sP(64x256) @ W_eout(256x64), +bias, *em ---
#pragma unroll
        for (int u = 0; u < 2; ++u) {
            int t = wv * 2 + u, ms = t >> 2, nt2 = t & 3;
            v8f acc = {};
#pragma unroll
            for (int kt = 0; kt < 8; ++kt) {
                v16bf af = *(const v16bf*)(sP + ((ms * 8 + kt) * 32 + lane) * 16);
                v16bf bf = *(const v16bf*)(pWeo + ((nt2 * 8 + kt) * 32 + lane) * 16);
                acc = __builtin_amdgcn_wmma_f32_16x16x32_bf16(false, af, false, bf, (short)0, acc, false, false);
            }
            int co = nt2 * 16 + l16;
            float bo = beo[co];
#pragma unroll
            for (int r = 0; r < 8; ++r) {
                int j = jb * 64 + ms * 16 + lh * 8 + r;
                float em = xmi * smask[j];
                outE[(((size_t)b * NN + i) * NN + j) * DEC + co] = (acc[r] + bo) * em;
            }
        }
        __syncthreads();
    }

    // --- cross-wave softmax combine: 8 partials per channel ---
#pragma unroll
    for (int tt = 0; tt < 8; ++tt) {
        int ch = (nhalf * 8 + tt) * 16 + l16;
        int s = ch * 8 + stripe * 2 + lh;
        sStat[s] = st_m[tt]; sStat[2048 + s] = st_l[tt]; sStat[4096 + s] = st_a[tt];
    }
    __syncthreads();
    {
        int ch = tid;
        float m = NEGBIG, l = 0.f, a = 0.f;
#pragma unroll
        for (int p = 0; p < 8; ++p) {
            float mp = sStat[ch * 8 + p], lp = sStat[2048 + ch * 8 + p], ap = sStat[4096 + ch * 8 + p];
            if (lp > 0.f) {
                if (l == 0.f) { m = mp; l = lp; a = ap; }
                else {
                    float mn2 = fmaxf(m, mp);
                    float s1 = __expf(m - mn2), s2 = __expf(mp - mn2);
                    l = l * s1 + lp * s2; a = a * s1 + ap * s2; m = mn2;
                }
            }
        }
        wVo[((size_t)b * NN + i) * DXC + ch] = a / l;
    }

    // --- fused E pooling atomics (per channel) ---
    atomicAdd(&zsum[b * DEC + pc], psum);
    atomicAdd(&zsq[b * DEC + pc], psq);
    atomicMin(&zminE[b * DEC + pc], fenc(pmn));
    atomicMax(&zmaxE[b * DEC + pc], fenc(pmx));
}

// ---------------------------------------------------------------------------
// newX = (yX_add + (yX_mul+1)*wV) @ W_xout + b, * xm   (WMMA)
// ---------------------------------------------------------------------------
__global__ __launch_bounds__(256) void newx_kernel(
    const float* __restrict__ wV, const int* __restrict__ nm,
    const float* __restrict__ yXm, const float* __restrict__ yXa,
    const __bf16* __restrict__ pWxo, const float* __restrict__ bxo, float* __restrict__ outX)
{
    __shared__ __align__(32) __bf16 sA[64 * 256];
    const int row0 = blockIdx.x * 64, b = row0 >> 7, tid = threadIdx.x;
    for (int e = tid; e < 64 * 256; e += 256) {
        int r = e >> 8, c = e & 255;
        float pre = yXa[b * DXC + c] + (yXm[b * DXC + c] + 1.0f) * wV[(size_t)(row0 + r) * DXC + c];
        sA[frag_off(8, r >> 4, c >> 5, r & 15, c & 31)] = (__bf16)pre;
    }
    __syncthreads();
    const int wv = tid >> 5, lane = tid & 31, lh = lane >> 4, l16 = lane & 15;
    for (int t = wv; t < 64; t += 8) {
        int ms = t >> 4, nt = t & 15;
        v8f acc = {};
#pragma unroll
        for (int kt = 0; kt < 8; ++kt) {
            v16bf af = *(const v16bf*)(sA + ((ms * 8 + kt) * 32 + lane) * 16);
            v16bf bf = *(const v16bf*)(pWxo + ((nt * 8 + kt) * 32 + lane) * 16);
            acc = __builtin_amdgcn_wmma_f32_16x16x32_bf16(false, af, false, bf, (short)0, acc, false, false);
        }
        int c = nt * 16 + l16;
        float bb = bxo[c];
#pragma unroll
        for (int r = 0; r < 8; ++r) {
            int row = row0 + ms * 16 + lh * 8 + r;
            outX[(size_t)row * DXC + c] = (acc[r] + bb) * (float)nm[row];
        }
    }
}

// ---------------------------------------------------------------------------
// new_y MLP (tiny): new_y = y@Wyy + zx@Wxy + ze@Wey + biases; relu mlp.
// ---------------------------------------------------------------------------
__global__ void y_final(const float* __restrict__ y, const float* __restrict__ zx,
                        const float* zsum, const float* zsq,
                        const unsigned* zmin, const unsigned* zmax,
                        const float* Wyy, const float* byy, const float* Wxy, const float* bxy,
                        const float* Wey, const float* bey,
                        const float* Wy1, const float* by1, const float* Wy2, const float* by2,
                        float* __restrict__ outY)
{
    __shared__ float sy[64], sze[256], szx[1024], sny[64], sh[64];
    int b = blockIdx.x, t = threadIdx.x;
    sy[t] = y[b * 64 + t];
    for (int k = t; k < 1024; k += 64) szx[k] = zx[b * 1024 + k];
    float mean = zsum[b * 64 + t] / 16384.f;
    float var = fmaxf((zsq[b * 64 + t] - 16384.f * mean * mean) / 16383.f, 0.f);
    sze[t] = mean;
    sze[64 + t] = fdec(zmin[b * 64 + t]);
    sze[128 + t] = fdec(zmax[b * 64 + t]);
    sze[192 + t] = sqrtf(var);
    __syncthreads();
    float acc = byy[t] + bxy[t] + bey[t];
    for (int k = 0; k < 64; ++k)   acc += sy[k] * Wyy[k * 64 + t];
    for (int k = 0; k < 1024; ++k) acc += szx[k] * Wxy[k * 64 + t];
    for (int k = 0; k < 256; ++k)  acc += sze[k] * Wey[k * 64 + t];
    sny[t] = acc;
    __syncthreads();
    float h = by1[t];
    for (int k = 0; k < 64; ++k) h += sny[k] * Wy1[k * 64 + t];
    h = fmaxf(h, 0.f);
    sh[t] = h;
    __syncthreads();
    float o = by2[t];
    for (int k = 0; k < 64; ++k) o += sh[k] * Wy2[k * 64 + t];
    outY[b * 64 + t] = o;
}

// ---------------------------------------------------------------------------
extern "C" void kernel_launch(void* const* d_in, const int* in_sizes, int n_in,
                              void* d_out, int out_size, void* d_ws, size_t ws_size,
                              hipStream_t stream) {
    (void)in_sizes; (void)n_in; (void)out_size; (void)ws_size;
    const float* X   = (const float*)d_in[0];
    const float* E   = (const float*)d_in[1];
    const float* y   = (const float*)d_in[2];
    const int*   nm  = (const int*)d_in[3];
    const float* Wq  = (const float*)d_in[4],  *bq  = (const float*)d_in[5];
    const float* Wk  = (const float*)d_in[6],  *bk  = (const float*)d_in[7];
    const float* Wv  = (const float*)d_in[8],  *bv  = (const float*)d_in[9];
    const float* Wea = (const float*)d_in[10], *bea = (const float*)d_in[11]; // e_add
    const float* Wem = (const float*)d_in[12], *bem = (const float*)d_in[13]; // e_mul
    const float* Wyem= (const float*)d_in[14], *byem= (const float*)d_in[15];
    const float* Wyea= (const float*)d_in[16], *byea= (const float*)d_in[17];
    const float* Wyxm= (const float*)d_in[18], *byxm= (const float*)d_in[19];
    const float* Wyxa= (const float*)d_in[20], *byxa= (const float*)d_in[21];
    const float* Wyy = (const float*)d_in[22], *byy = (const float*)d_in[23];
    const float* Wxy = (const float*)d_in[24], *bxy = (const float*)d_in[25];
    const float* Wey = (const float*)d_in[26], *bey = (const float*)d_in[27];
    const float* Wxo = (const float*)d_in[28], *bxo = (const float*)d_in[29];
    const float* Weo = (const float*)d_in[30], *beo = (const float*)d_in[31];
    const float* Wy1 = (const float*)d_in[32], *by1 = (const float*)d_in[33];
    const float* Wy2 = (const float*)d_in[34], *by2 = (const float*)d_in[35];

    float* ws   = (float*)d_ws;
    float* Qb   = ws;                 // 2048*256
    float* Kb   = Qb + 524288;
    float* Vb   = Kb + 524288;
    float* wV   = Vb + 524288;
    float* yEm  = wV + 524288;        // 16*256 each
    float* yEa  = yEm + 4096;
    float* yXm  = yEa + 4096;
    float* yXa  = yXm + 4096;
    float* zx   = yXa + 4096;         // 16*1024
    float* zsum = zx + 16384;         // 16*64 each
    float* zsq  = zsum + 1024;
    unsigned* zmin = (unsigned*)(zsq + 1024);
    unsigned* zmax = zmin + 1024;
    __bf16* pWq  = (__bf16*)(zmax + 1024);
    __bf16* pWk  = pWq  + 65536;
    __bf16* pWv  = pWk  + 65536;
    __bf16* pWem = pWv  + 65536;
    __bf16* pWea = pWem + 16384;
    __bf16* pWeo = pWea + 16384;
    __bf16* pWxo = pWeo + 16384;

    float* outX = (float*)d_out;
    float* outE = outX + 524288;
    float* outY = outE + 16777216;

    init_stats<<<4, 256, 0, stream>>>(zsum, zsq, zmin, zmax);
    pack_w<<<256, 256, 0, stream>>>(Wq,  pWq,  256, 256);
    pack_w<<<256, 256, 0, stream>>>(Wk,  pWk,  256, 256);
    pack_w<<<256, 256, 0, stream>>>(Wv,  pWv,  256, 256);
    pack_w<<<64,  256, 0, stream>>>(Wem, pWem, 64,  256);
    pack_w<<<64,  256, 0, stream>>>(Wea, pWea, 64,  256);
    pack_w<<<64,  256, 0, stream>>>(Weo, pWeo, 256, 64);
    pack_w<<<256, 256, 0, stream>>>(Wxo, pWxo, 256, 256);

    ymix<<<BSZ, 256, 0, stream>>>(y, Wyem, byem, Wyea, byea, Wyxm, byxm, Wyxa, byxa,
                                  yEm, yEa, yXm, yXa);
    node_proj<<<32, 256, 0, stream>>>(X, nm, pWq, pWk, pWv, bq, bk, bv, Qb, Kb, Vb);
    x_pool<<<BSZ, 256, 0, stream>>>(X, zx);

    edge_kernel<<<BSZ * NN, 256, 0, stream>>>(E, nm, Qb, Kb, Vb, yEm, yEa,
                                              pWem, pWea, pWeo, bem, bea, beo,
                                              outE, wV, zsum, zsq, zmin, zmax);

    newx_kernel<<<32, 256, 0, stream>>>(wV, nm, yXm, yXa, pWxo, bxo, outX);

    y_final<<<BSZ, 64, 0, stream>>>(y, zx, zsum, zsq, zmin, zmax,
                                    Wyy, byy, Wxy, bxy, Wey, bey,
                                    Wy1, by1, Wy2, by2, outY);
}
